// UncertaintyAttentionModulator_27376121545123
// MI455X (gfx1250) — matrix operand order repaired and checked by reading, not compiled, not run on hardware
//
#include <hip/hip_runtime.h>
#include <hip/hip_bf16.h>

// ---------- CDNA5 WMMA / TDM types ----------
typedef _Float16 h8   __attribute__((ext_vector_type(8)));
typedef _Float16 v16h __attribute__((ext_vector_type(16)));
typedef float    v8f  __attribute__((ext_vector_type(8)));
typedef unsigned int v4u __attribute__((ext_vector_type(4)));
typedef int          v8i __attribute__((ext_vector_type(8)));
typedef int          v4i __attribute__((ext_vector_type(4)));

union Frag { v16h v; h8 h[2]; };

static __device__ __forceinline__ v8f wmma_f16(v16h a, v16h b, v8f c) {
  // D = A(16x32 f16) * B(32x16 f16) + C(16x16 f32)
  return __builtin_amdgcn_wmma_f32_16x16x32_f16(false, a, false, b, (short)0, c,
                                                false, false);
}

static __device__ __forceinline__ h8 cvt8(float4 a, float4 b) {
  h8 r;
  r[0] = (_Float16)a.x; r[1] = (_Float16)a.y; r[2] = (_Float16)a.z; r[3] = (_Float16)a.w;
  r[4] = (_Float16)b.x; r[5] = (_Float16)b.y; r[6] = (_Float16)b.z; r[7] = (_Float16)b.w;
  return r;
}

static __device__ __forceinline__ unsigned ldsOff(const void* p) {
  // low 32 bits of a generic pointer to LDS == LDS byte offset
  return (unsigned)(unsigned long long)p;
}

// Tensor Data Mover: async 2D tile load global -> LDS (D# per ISA 8.3/8.4)
// tile_d0 = contiguous elements per row, tile_d1 = rows, d0_stride = row stride
// pad_icode/pad_acode: LDS padding (interval code, amount code); data_size = 4B.
// This toolchain uses the 6-arg builtin: (g0, g1, g2, g3, g4, cpol).
static __device__ __forceinline__ void tdm_load_2d_f32(
    unsigned lds_addr, const float* gptr,
    unsigned tensor_d0, unsigned tensor_d1, unsigned d0_stride,
    unsigned tile_d0, unsigned tile_d1,
    unsigned pad_icode, unsigned pad_acode) {
  unsigned long long ga = (unsigned long long)(const void*)gptr;
  v4u g0;
  g0[0] = 1u;                                   // count=1, user mode, no gather
  g0[1] = lds_addr;                             // lds_addr [63:32]
  g0[2] = (unsigned)(ga & 0xFFFFFFFFu);         // global_addr lo
  g0[3] = (unsigned)((ga >> 32) & 0x01FFFFFFu)  // global_addr hi (bits 56:32)
          | (2u << 30);                         // type = 2 ("image")
  v8i g1;
  unsigned w0 = (2u << 16)                      // data_size = 4 bytes
              | (1u << 20)                      // pad_enable
              | (pad_icode << 22) | (pad_acode << 25);
  g1[0] = (int)w0;
  g1[1] = (int)((tensor_d0 & 0xFFFFu) << 16);                       // dim0 lo16
  g1[2] = (int)(((tensor_d0 >> 16) & 0xFFFFu)                       // dim0 hi16
          | ((tensor_d1 & 0xFFFFu) << 16));                         // dim1 lo16
  g1[3] = (int)(((tensor_d1 >> 16) & 0xFFFFu) | (tile_d0 << 16));   // tile_dim0
  g1[4] = (int)(tile_d1 & 0xFFFFu);                                 // tile_dim1
  g1[5] = (int)d0_stride;                                           // stride lo32
  g1[6] = 0;                                                        // stride hi
  g1[7] = 0;
  v4i z4 = {};
  v8i z8 = {};
  __builtin_amdgcn_tensor_load_to_lds(g0, g1, z4, z4, z8, 0);
}

// ---------- problem constants ----------
#define HIDDEN 1024
#define HEADS  16
#define HDIM   64
#define BB     2
#define LL     2048
#define MROWS  (BB * LL)   // 4096

// =====================================================================
// Kernel 1: uncertainty gate  scale[b,h] = sigmoid(relu(ue@Wu1+bu1)@Wu2+bu2)
// =====================================================================
__global__ __launch_bounds__(256) void uncert_kernel(
    const float* __restrict__ ue, const float* __restrict__ Wu1,
    const float* __restrict__ bu1, const float* __restrict__ Wu2,
    const float* __restrict__ bu2, float* __restrict__ scale) {
  __shared__ float su[256];
  const int b = blockIdx.x, t = threadIdx.x;
  const float* e = ue + (size_t)b * HIDDEN;
  float acc = bu1[t];
  for (int k = 0; k < HIDDEN; ++k) acc += e[k] * Wu1[k * 256 + t];
  su[t] = fmaxf(acc, 0.0f);
  __syncthreads();
  if (t < HEADS) {
    float a2 = bu2[t];
    for (int k = 0; k < 256; ++k) a2 += su[k] * Wu2[k * HEADS + t];
    scale[b * HEADS + t] = 1.0f / (1.0f + __expf(-a2));
  }
}

// =====================================================================
// Kernel 2: C[M,N] = A[M,K] @ W[K,N] + bias
// TDM double-buffered fp32 tiles in LDS, f16 WMMA compute.
// block tile 128x128, 8 waves, wave tile 32x64, K-step 32.
// All 6 fragments materialized before the 8 WMMAs to avoid WMMA<->VALU
// WAR hazard nops on shared operand registers.
// =====================================================================
__global__ __launch_bounds__(256) void gemm_bias_kernel(
    const float* __restrict__ A, const float* __restrict__ W,
    const float* __restrict__ bias, float* __restrict__ C,
    int M, int K, int N) {
  __shared__ float sAf[2][128 * 36];
  __shared__ float sWf[2][32 * 132];
  const int t = threadIdx.x;
  const int lane = t & 31, wv = t >> 5;
  const int m0 = blockIdx.x * 128;
  const int n0 = blockIdx.y * 128;
  const int r0 = (wv & 3) * 32;
  const int c0 = (wv >> 2) * 64;

  v8f acc[2][4];
  v8f zz = {};
  for (int ri = 0; ri < 2; ++ri)
    for (int s = 0; s < 4; ++s) acc[ri][s] = zz;

  const int kA = (lane >> 4) * 8;   // A-fragment half-wave K offset

  // prologue: DMA tile 0 into buffer 0 (one wave issues; TDM ignores EXEC)
  if (wv == 0) {
    tdm_load_2d_f32(ldsOff(&sAf[0][0]), A + (size_t)m0 * K,
                    (unsigned)K, (unsigned)M, (unsigned)K, 32u, 128u, 4u, 3u);
    tdm_load_2d_f32(ldsOff(&sWf[0][0]), W + n0,
                    (unsigned)N, (unsigned)K, (unsigned)N, 128u, 32u, 6u, 3u);
  }

  for (int k0 = 0; k0 < K; k0 += 32) {
    const int bsel = (k0 >> 5) & 1;
    if (wv == 0) __builtin_amdgcn_s_wait_tensorcnt(0);
    __syncthreads();   // publish tile k; all waves done reading other buffer
    if (wv == 0 && k0 + 32 < K) {
      tdm_load_2d_f32(ldsOff(&sAf[bsel ^ 1][0]), A + (size_t)m0 * K + (k0 + 32),
                      (unsigned)K, (unsigned)M, (unsigned)K, 32u, 128u, 4u, 3u);
      tdm_load_2d_f32(ldsOff(&sWf[bsel ^ 1][0]),
                      W + (size_t)(k0 + 32) * N + n0,
                      (unsigned)N, (unsigned)K, (unsigned)N, 128u, 32u, 6u, 3u);
    }
    const float* sAc = &sAf[bsel][0];
    const float* sWc = &sWf[bsel][0];

    // materialize ALL fragments for this K-step (distinct registers)
    Frag a[2];
    for (int ri = 0; ri < 2; ++ri) {
      const float* base = sAc + (r0 + ri * 16 + (lane & 15)) * 36 + kA;
      float4 f0 = *(const float4*)(base);
      float4 f1 = *(const float4*)(base + 4);
      float4 f2 = *(const float4*)(base + 16);
      float4 f3 = *(const float4*)(base + 20);
      a[ri].h[0] = cvt8(f0, f1);
      a[ri].h[1] = cvt8(f2, f3);
    }
    Frag bf[4];
    for (int s = 0; s < 4; ++s) {
      const float* base = sWc + kA * 132 + c0 + s * 16 + (lane & 15);
      for (int j = 0; j < 8; ++j) {
        bf[s].h[0][j] = (_Float16)base[j * 132];
        bf[s].h[1][j] = (_Float16)base[(j + 16) * 132];
      }
    }
    // 8 back-to-back WMMAs, no VALU writes to their source registers
    for (int s = 0; s < 4; ++s)
      for (int ri = 0; ri < 2; ++ri)
        acc[ri][s] = wmma_f16(a[ri].v, bf[s].v, acc[ri][s]);
  }

  // epilogue: bias + store (C/D layout: n = lane&15, m = elem + 8*(lane>>4))
  for (int ri = 0; ri < 2; ++ri)
    for (int s = 0; s < 4; ++s) {
      int col = n0 + c0 + s * 16 + (lane & 15);
      float bv = bias[col];
      int rbase = m0 + r0 + ri * 16 + 8 * (lane >> 4);
      for (int i = 0; i < 8; ++i)
        C[(size_t)(rbase + i) * N + col] = acc[ri][s][i] + bv;
    }
}

// =====================================================================
// Kernel 3: flash attention per (b,h); block = 128 queries, 8 waves x 16q
// key tiles of 64; online softmax; WMMA for QK^T and PV
// =====================================================================
__global__ __launch_bounds__(256) void attn_kernel(
    const float* __restrict__ Q, const float* __restrict__ Kb,
    const float* __restrict__ V, const int* __restrict__ mask,
    const float* __restrict__ scale, float* __restrict__ Oout) {
  __shared__ _Float16 sK[64 * 72];        // K^T tile: [d][key], stride 72
  __shared__ _Float16 sV[64 * 72];        // V tile:   [key][d], stride 72
  __shared__ _Float16 sP[8 * 16 * 72];    // per-wave P: [m][key]
  __shared__ int sM[64];

  const int t = threadIdx.x, lane = t & 31, wv = t >> 5;
  const int QT = LL / 128;
  const int bh = blockIdx.x / QT, qt = blockIdx.x % QT;
  const int b = bh >> 4, h = bh & (HEADS - 1);
  const int q0 = qt * 128 + wv * 16;

  const float hscale = scale[bh];
  const float sc = hscale * 0.125f;       // scale / sqrt(64)

  // Q fragments (persistent): 16 queries x 64 d, two 32-wide K-chunks
  Frag qa[2];
  {
    const int m = lane & 15, kg = (lane >> 4) * 8;
    const float* qrow = Q + ((size_t)(b * LL + q0 + m)) * HIDDEN + h * HDIM;
    for (int c = 0; c < 2; ++c) {
      float4 f0 = *(const float4*)(qrow + c * 32 + kg);
      float4 f1 = *(const float4*)(qrow + c * 32 + kg + 4);
      float4 f2 = *(const float4*)(qrow + c * 32 + kg + 16);
      float4 f3 = *(const float4*)(qrow + c * 32 + kg + 20);
      qa[c].h[0] = cvt8(f0, f1);
      qa[c].h[1] = cvt8(f2, f3);
    }
  }

  float mrun[8], lrun[8];
  v8f Oa[4];
  v8f zz = {};
  for (int i = 0; i < 8; ++i) { mrun[i] = -1.0e30f; lrun[i] = 0.0f; }
  for (int i = 0; i < 4; ++i) Oa[i] = zz;

  for (int kt0 = 0; kt0 < LL; kt0 += 64) {
    __syncthreads();
    // stage K (transposed) and V tiles: batch all global loads first
    float4 rk[4], rv[4];
    int keyi[4], d4i[4];
    for (int i = 0; i < 4; ++i) {
      int idx = t + i * 256;
      keyi[i] = idx >> 4;
      d4i[i] = (idx & 15) * 4;
      size_t gofs = ((size_t)(b * LL + kt0 + keyi[i])) * HIDDEN + h * HDIM + d4i[i];
      rk[i] = *(const float4*)(Kb + gofs);
      rv[i] = *(const float4*)(V + gofs);
    }
    for (int i = 0; i < 4; ++i) {
      int key = keyi[i], d4 = d4i[i];
      sK[(d4 + 0) * 72 + key] = (_Float16)rk[i].x;
      sK[(d4 + 1) * 72 + key] = (_Float16)rk[i].y;
      sK[(d4 + 2) * 72 + key] = (_Float16)rk[i].z;
      sK[(d4 + 3) * 72 + key] = (_Float16)rk[i].w;
      _Float16* vd = &sV[key * 72 + d4];
      vd[0] = (_Float16)rv[i].x; vd[1] = (_Float16)rv[i].y;
      vd[2] = (_Float16)rv[i].z; vd[3] = (_Float16)rv[i].w;
    }
    if (t < 64) sM[t] = mask[b * LL + kt0 + t];
    __syncthreads();

    // S = Q @ K^T  (4 key subtiles x 2 d-chunks)
    v8f S[4];
    for (int s = 0; s < 4; ++s) {
      v8f c = zz;
      for (int dc = 0; dc < 2; ++dc) {
        Frag bf;
        const _Float16* base = &sK[(dc * 32 + (lane >> 4) * 8) * 72 + s * 16 + (lane & 15)];
        for (int j = 0; j < 8; ++j) {
          bf.h[0][j] = base[j * 72];
          bf.h[1][j] = base[(j + 16) * 72];
        }
        c = wmma_f16(qa[dc].v, bf.v, c);
      }
      S[s] = c;
    }

    // scale + mask + tile row-max
    float tmax[8];
    for (int i = 0; i < 8; ++i) tmax[i] = -3.0e38f;
    for (int s = 0; s < 4; ++s) {
      bool ok = sM[s * 16 + (lane & 15)] != 0;
      for (int i = 0; i < 8; ++i) {
        float v = ok ? S[s][i] * sc : -1.0e9f * hscale;
        S[s][i] = v;
        tmax[i] = fmaxf(tmax[i], v);
      }
    }
    for (int off = 1; off < 16; off <<= 1)
      for (int i = 0; i < 8; ++i)
        tmax[i] = fmaxf(tmax[i], __shfl_xor(tmax[i], off, 32));

    // online softmax update
    float rsum[8];
    for (int i = 0; i < 8; ++i) {
      float mn = fmaxf(mrun[i], tmax[i]);
      float alpha = __expf(mrun[i] - mn);
      mrun[i] = mn;
      lrun[i] *= alpha;
      for (int tt = 0; tt < 4; ++tt) Oa[tt][i] *= alpha;
      rsum[i] = 0.0f;
    }
    for (int s = 0; s < 4; ++s)
      for (int i = 0; i < 8; ++i) {
        float p = __expf(S[s][i] - mrun[i]);
        S[s][i] = p;
        rsum[i] += p;
      }
    for (int off = 1; off < 16; off <<= 1)
      for (int i = 0; i < 8; ++i) rsum[i] += __shfl_xor(rsum[i], off, 32);
    for (int i = 0; i < 8; ++i) lrun[i] += rsum[i];

    // write P (D-layout) to per-wave LDS, reload in A-layout
    _Float16* pw = &sP[wv * 16 * 72];
    for (int s = 0; s < 4; ++s) {
      int key = s * 16 + (lane & 15);
      for (int i = 0; i < 8; ++i) {
        int m = i + 8 * (lane >> 4);
        pw[m * 72 + key] = (_Float16)S[s][i];
      }
    }
    __syncthreads();

    Frag pf[2];
    {
      const int m = lane & 15, kg = (lane >> 4) * 8;
      for (int c = 0; c < 2; ++c) {
        const _Float16* base = &pw[m * 72 + c * 32 + kg];
        pf[c].h[0] = *(const h8*)(base);
        pf[c].h[1] = *(const h8*)(base + 16);
      }
    }
    // O += P @ V
    for (int tt = 0; tt < 4; ++tt)
      for (int c = 0; c < 2; ++c) {
        Frag vf;
        const _Float16* base = &sV[(c * 32 + (lane >> 4) * 8) * 72 + tt * 16 + (lane & 15)];
        for (int j = 0; j < 8; ++j) {
          vf.h[0][j] = base[j * 72];
          vf.h[1][j] = base[(j + 16) * 72];
        }
        Oa[tt] = wmma_f16(pf[c].v, vf.v, Oa[tt]);
      }
  }

  // normalize + store (merge heads back into (B,L,HIDDEN))
  for (int tt = 0; tt < 4; ++tt) {
    int d = h * HDIM + tt * 16 + (lane & 15);
    for (int i = 0; i < 8; ++i) {
      int q = q0 + i + 8 * (lane >> 4);
      Oout[((size_t)(b * LL + q)) * HIDDEN + d] = Oa[tt][i] / lrun[i];
    }
  }
}

// =====================================================================
// Kernel 4: residual + LayerNorm
// =====================================================================
__global__ __launch_bounds__(256) void ln_kernel(
    const float* __restrict__ hidden, const float* __restrict__ proj,
    const float* __restrict__ gamma, const float* __restrict__ beta,
    float* __restrict__ out) {
  __shared__ float ssum[256], ssq[256];
  const int row = blockIdx.x, t = threadIdx.x;
  const float* hp = hidden + (size_t)row * HIDDEN;
  const float* pp = proj + (size_t)row * HIDDEN;
  float y[4], s = 0.0f, q = 0.0f;
  for (int i = 0; i < 4; ++i) {
    float v = hp[t + i * 256] + pp[t + i * 256];
    y[i] = v; s += v; q += v * v;
  }
  ssum[t] = s; ssq[t] = q;
  __syncthreads();
  for (int off = 128; off > 0; off >>= 1) {
    if (t < off) { ssum[t] += ssum[t + off]; ssq[t] += ssq[t + off]; }
    __syncthreads();
  }
  float mean = ssum[0] * (1.0f / HIDDEN);
  float var  = ssq[0] * (1.0f / HIDDEN) - mean * mean;
  float inv  = rsqrtf(var + 1e-5f);
  for (int i = 0; i < 4; ++i) {
    int c = t + i * 256;
    out[(size_t)row * HIDDEN + c] = (y[i] - mean) * inv * gamma[c] + beta[c];
  }
}

// =====================================================================
// launch
// =====================================================================
extern "C" void kernel_launch(void* const* d_in, const int* in_sizes, int n_in,
                              void* d_out, int out_size, void* d_ws, size_t ws_size,
                              hipStream_t stream) {
  (void)in_sizes; (void)n_in; (void)out_size; (void)ws_size;
  const float* hidden = (const float*)d_in[0];
  const float* ue     = (const float*)d_in[1];
  const int*   amask  = (const int*)  d_in[2];
  const float* Wq = (const float*)d_in[3];  const float* bq = (const float*)d_in[4];
  const float* Wk = (const float*)d_in[5];  const float* bk = (const float*)d_in[6];
  const float* Wv = (const float*)d_in[7];  const float* bv = (const float*)d_in[8];
  const float* Wo = (const float*)d_in[9];  const float* bo = (const float*)d_in[10];
  const float* Wu1 = (const float*)d_in[11]; const float* bu1 = (const float*)d_in[12];
  const float* Wu2 = (const float*)d_in[13]; const float* bu2 = (const float*)d_in[14];
  const float* gamma = (const float*)d_in[15]; const float* beta = (const float*)d_in[16];
  float* out = (float*)d_out;

  const size_t SZ = (size_t)MROWS * HIDDEN;
  float* ws   = (float*)d_ws;
  float* wsQ  = ws;            // also reused for output projection result
  float* wsK  = ws + SZ;
  float* wsV  = ws + 2 * SZ;
  float* wsA  = ws + 3 * SZ;
  float* wsSc = ws + 4 * SZ;   // 32 floats: scale[b,h]

  uncert_kernel<<<BB, 256, 0, stream>>>(ue, Wu1, bu1, Wu2, bu2, wsSc);

  dim3 gg(MROWS / 128, HIDDEN / 128);
  gemm_bias_kernel<<<gg, 256, 0, stream>>>(hidden, Wq, bq, wsQ, MROWS, HIDDEN, HIDDEN);
  gemm_bias_kernel<<<gg, 256, 0, stream>>>(hidden, Wk, bk, wsK, MROWS, HIDDEN, HIDDEN);
  gemm_bias_kernel<<<gg, 256, 0, stream>>>(hidden, Wv, bv, wsV, MROWS, HIDDEN, HIDDEN);

  attn_kernel<<<BB * HEADS * (LL / 128), 256, 0, stream>>>(wsQ, wsK, wsV, amask, wsSc, wsA);

  gemm_bias_kernel<<<gg, 256, 0, stream>>>(wsA, Wo, bo, wsQ, MROWS, HIDDEN, HIDDEN);

  ln_kernel<<<MROWS, 256, 0, stream>>>(hidden, wsQ, gamma, beta, out);
}
